// SinkhornAttention_87033217286667
// MI455X (gfx1250) — matrix operand order repaired and checked by reading, not compiled, not run on hardware
//
#include <hip/hip_runtime.h>
#include <hip/hip_bf16.h>
#include <math.h>

// ---------------------------------------------------------------------------
// Sinkhorn bucket attention for MI455X (gfx1250, wave32, WMMA).
// Shapes: b=2 h=8 t=8192 dh=128 -> bh=16, BUCKETS=128, BSZ=64.
// HBM-bound (~350MB @ 23.3TB/s ~ 15us); compute (8.6 GFLOP) done with
// v_wmma_f32_16x16x32_bf16 (f32 accumulate) out of LDS-staged tiles.
// ---------------------------------------------------------------------------

#define BH        16
#define T_LEN     8192
#define DH        128
#define BUCKETS   128
#define BSZ       64
#define EPS_F     1e-6f
#define INV_TEMP  (1.0f / 0.75f)
#define SM_SCALE  0.03125f          // (h*dh)^-0.5 = 1024^-0.5

typedef __attribute__((ext_vector_type(16))) __bf16 v16bf;
typedef __attribute__((ext_vector_type(8)))  __bf16 v8bf;
typedef __attribute__((ext_vector_type(8)))  float  v8f;

// LDS row strides (in elements). 136 bf16 = 272B = 17*16B -> every row start
// and every 8-element column offset is 16B aligned for ds_load_b128, and the
// 68-dword stride breaks the worst bank-conflict patterns.
#define LDK 136   // bf16 matrices (sq / sk / svT)
#define LDA 132   // f32 scores (132*4 = 528 = 33*16B)

// A-fragment (16x32 bf16, MxK): lane holds one row; lanes 0-15 take
// K = {kc..kc+7, kc+16..kc+23}, lanes 16-31 take K = {kc+8..15, kc+24..31}.
// Caller passes p = &row[kc + hi*8]; we grab p[0..7] and p[16..23].
__device__ __forceinline__ v16bf frag_a(const __bf16* p) {
  v8bf lo = *(const v8bf*)(p);
  v8bf hh = *(const v8bf*)(p + 16);
  return __builtin_shufflevector(lo, hh, 0,1,2,3,4,5,6,7,8,9,10,11,12,13,14,15);
}
// B-fragment (32x16 bf16, KxN): lane holds one column; lanes 0-15 take
// K = {kc..kc+7, kc+16..kc+23}, lanes 16-31 take K = {kc+8..15, kc+24..31}.
__device__ __forceinline__ v16bf frag_b(const __bf16* p) {
  v8bf lo = *(const v8bf*)(p);
  v8bf hh = *(const v8bf*)(p + 16);
  return __builtin_shufflevector(lo, hh, 0,1,2,3,4,5,6,7,8,9,10,11,12,13,14,15);
}

// ---------------------------------------------------------------------------
// Kernel 1: SortNet. One block per (bucket u, bh). Fuses the per-bucket q/k
// sums, the 256->128 linear, relu, and the gumbel/log transform
// r = (log(relu(xW)+eps) + g) / T.  Writes r to ws.
// ---------------------------------------------------------------------------
__global__ __launch_bounds__(128) void sortnet_kernel(
    const float* __restrict__ q, const float* __restrict__ k,
    const float* __restrict__ W, const float* __restrict__ ug,
    float* __restrict__ rbuf) {
  const int u   = blockIdx.x;
  const int bh  = blockIdx.y;
  const int tid = threadIdx.x;          // 0..127 == feature / out column

  __shared__ float xrow[256];

  const float* qb = q + ((size_t)bh * T_LEN + (size_t)u * BSZ) * DH;
  const float* kb = k + ((size_t)bh * T_LEN + (size_t)u * BSZ) * DH;
  float sq = 0.f, sk = 0.f;
  #pragma unroll 4
  for (int i = 0; i < BSZ; ++i) {
    sq += qb[i * DH + tid];
    sk += kb[i * DH + tid];
  }
  xrow[tid]       = sq;
  xrow[128 + tid] = sk;
  __syncthreads();

  const int h = bh & 7;                 // W_sort broadcast over batch
  const float* Wc = W + (size_t)h * 256 * BUCKETS;
  float acc = 0.f;
  #pragma unroll 8
  for (int e = 0; e < 256; ++e)
    acc = fmaf(xrow[e], Wc[e * BUCKETS + tid], acc);

  const float R  = fmaxf(acc, 0.f);
  const float uu = ug[((size_t)bh * BUCKETS + u) * BUCKETS + tid];
  const float g  = -__logf(-__logf(uu + EPS_F) + EPS_F);
  rbuf[((size_t)bh * BUCKETS + u) * BUCKETS + tid] =
      (__logf(R + EPS_F) + g) * INV_TEMP;
}

// ---------------------------------------------------------------------------
// Kernel 2: log-domain Sinkhorn (7 iters) + per-row top-1 -> (idx, exp(val)).
// One block per bh; 128x128 tile in LDS with stride 129 (129 mod 64 == 1, so
// both row and column sweeps are bank-conflict free).
// ---------------------------------------------------------------------------
__global__ __launch_bounds__(128) void sinkhorn_kernel(
    const float* __restrict__ rbuf, int* __restrict__ rowIdx,
    float* __restrict__ rowScale) {
  extern __shared__ float Rl[];         // 128 * 129 floats
  const int bh = blockIdx.x;
  const int t  = threadIdx.x;

  const float* src = rbuf + (size_t)bh * BUCKETS * BUCKETS;
  for (int uu = 0; uu < BUCKETS; ++uu)
    Rl[uu * 129 + t] = src[uu * BUCKETS + t];
  __syncthreads();

  for (int it = 0; it < 7; ++it) {
    // axis=2 (per row u = t)
    {
      float m = -3.4e38f;
      for (int vv = 0; vv < BUCKETS; ++vv) m = fmaxf(m, Rl[t * 129 + vv]);
      float s = 0.f;
      for (int vv = 0; vv < BUCKETS; ++vv) s += __expf(Rl[t * 129 + vv] - m);
      const float lse = m + __logf(s);
      for (int vv = 0; vv < BUCKETS; ++vv) Rl[t * 129 + vv] -= lse;
    }
    __syncthreads();
    // axis=1 (per column v = t)
    {
      float m = -3.4e38f;
      for (int uu = 0; uu < BUCKETS; ++uu) m = fmaxf(m, Rl[uu * 129 + t]);
      float s = 0.f;
      for (int uu = 0; uu < BUCKETS; ++uu) s += __expf(Rl[uu * 129 + t] - m);
      const float lse = m + __logf(s);
      for (int uu = 0; uu < BUCKETS; ++uu) Rl[uu * 129 + t] -= lse;
    }
    __syncthreads();
  }

  // top-1 per row (argmax in log domain == argmax after exp)
  float best = -3.4e38f; int bi = 0;
  for (int vv = 0; vv < BUCKETS; ++vv) {
    const float x = Rl[t * 129 + vv];
    if (x > best) { best = x; bi = vv; }
  }
  rowIdx[bh * BUCKETS + t]   = bi;
  rowScale[bh * BUCKETS + t] = __expf(best);
}

// ---------------------------------------------------------------------------
// Kernel 3: bucketed attention. One block per (u, bh); 256 threads = 8 waves.
// Stages q (bf16), k_cat (bf16), v_cat^T (bf16) in LDS, then:
//   GEMM1: dots(64x128) = q @ k_cat^T      (32 tiles, 4 WMMA bf16 each)
//   softmax rows in f32
//   GEMM2: out(64x128)  = attn @ v_cat     (32 tiles, 4 WMMA bf16 each)
// k_cat rows 0..63 are the top-1 gathered bucket scaled by R[u,v*].
// ---------------------------------------------------------------------------
__global__ __launch_bounds__(256) void attn_kernel(
    const float* __restrict__ q, const float* __restrict__ k,
    const float* __restrict__ v, const int* __restrict__ rowIdx,
    const float* __restrict__ rowScale, float* __restrict__ out) {
  extern __shared__ char smem[];
  __bf16* sq  = (__bf16*)smem;           // 64  x LDK  (q, later attn probs)
  __bf16* sk  = sq + 64 * LDK;           // 128 x LDK  (k_cat, row-major)
  __bf16* sv  = sk + 128 * LDK;          // 128 x LDK  (v_cat TRANSPOSED: [d][j])
  float*  sat = (float*)(sv + 128 * LDK);// 64  x LDA  (f32 scores)

  const int u   = blockIdx.x;
  const int bh  = blockIdx.y;
  const int tid = threadIdx.x;
  const int lane = tid & 31, wv = tid >> 5;
  const int hi = lane >> 4, ln = lane & 15;

  const int   vstar = rowIdx[bh * BUCKETS + u];
  const float scl   = rowScale[bh * BUCKETS + u];

  const size_t base   = (size_t)bh * T_LEN * DH;
  const float* qb     = q + base + (size_t)u * BSZ * DH;
  const float* kb_u   = k + base + (size_t)u * BSZ * DH;
  const float* kb_s   = k + base + (size_t)vstar * BSZ * DH;
  const float* vb_u   = v + base + (size_t)u * BSZ * DH;
  const float* vb_s   = v + base + (size_t)vstar * BSZ * DH;

  // stage q -> bf16
  for (int idx = tid; idx < 64 * 128; idx += 256) {
    const int r = idx >> 7, d = idx & 127;
    sq[r * LDK + d] = (__bf16)qb[r * DH + d];
  }
  // stage k_cat (row-major) and v_cat (transposed)
  for (int idx = tid; idx < 128 * 128; idx += 256) {
    const int j = idx >> 7, d = idx & 127;
    float kv, vv;
    if (j < 64) { kv = scl * kb_s[j * DH + d]; vv = scl * vb_s[j * DH + d]; }
    else        { kv = kb_u[(j - 64) * DH + d]; vv = vb_u[(j - 64) * DH + d]; }
    sk[j * LDK + d] = (__bf16)kv;
    sv[d * LDK + j] = (__bf16)vv;        // B-matrix wants v_cat columns per lane
  }
  __syncthreads();

  // ---- GEMM1: dots = q @ k_cat^T ------------------------------------------
  const int mt = wv >> 1;                // wave's M tile (4 tiles of 16 rows)
  #pragma unroll
  for (int i = 0; i < 4; ++i) {
    const int nt = (wv & 1) * 4 + i;     // N tile (8 tiles of 16 cols)
    v8f acc = {};
    #pragma unroll
    for (int kc = 0; kc < 128; kc += 32) {
      const v16bf a = frag_a(&sq[(mt * 16 + ln) * LDK + kc + hi * 8]);
      const v16bf b = frag_b(&sk[(nt * 16 + ln) * LDK + kc + hi * 8]);
      acc = __builtin_amdgcn_wmma_f32_16x16x32_bf16(
          false, a, false, b, (short)0, acc, false, false);
    }
    #pragma unroll
    for (int r = 0; r < 8; ++r)          // C layout: VGPR r -> row r / r+8
      sat[(mt * 16 + hi * 8 + r) * LDA + nt * 16 + ln] = acc[r] * SM_SCALE;
  }
  __syncthreads();

  // ---- softmax over 128 cols, write bf16 probs into (dead) sq -------------
  if (tid < 64) {
    float m = -3.4e38f;
    for (int j = 0; j < 128; ++j) m = fmaxf(m, sat[tid * LDA + j]);
    float s = 0.f;
    for (int j = 0; j < 128; ++j) s += __expf(sat[tid * LDA + j] - m);
    const float inv = 1.0f / s;
    for (int j = 0; j < 128; ++j)
      sq[tid * LDK + j] = (__bf16)(__expf(sat[tid * LDA + j] - m) * inv);
  }
  __syncthreads();

  // ---- GEMM2: out = attn @ v_cat (B fragments from transposed sv) ---------
  float* ob = out + ((size_t)bh * BUCKETS + u) * BSZ * DH;
  #pragma unroll
  for (int i = 0; i < 4; ++i) {
    const int dt = (wv & 1) * 4 + i;     // output d tile
    v8f acc = {};
    #pragma unroll
    for (int kc = 0; kc < 128; kc += 32) {
      const v16bf a = frag_a(&sq[(mt * 16 + ln) * LDK + kc + hi * 8]);
      const v16bf b = frag_b(&sv[(dt * 16 + ln) * LDK + kc + hi * 8]);
      acc = __builtin_amdgcn_wmma_f32_16x16x32_bf16(
          false, a, false, b, (short)0, acc, false, false);
    }
    #pragma unroll
    for (int r = 0; r < 8; ++r)
      ob[(mt * 16 + hi * 8 + r) * DH + dt * 16 + ln] = acc[r];
  }
}

// ---------------------------------------------------------------------------
extern "C" void kernel_launch(void* const* d_in, const int* in_sizes, int n_in,
                              void* d_out, int out_size, void* d_ws, size_t ws_size,
                              hipStream_t stream) {
  const float* q  = (const float*)d_in[0];
  const float* k  = (const float*)d_in[1];
  const float* v  = (const float*)d_in[2];
  const float* W  = (const float*)d_in[3];   // (1,8,256,128)
  const float* ug = (const float*)d_in[4];   // (16,128,128)
  float* out = (float*)d_out;

  // workspace layout
  float* rbuf     = (float*)d_ws;                                  // 16*128*128 f32 = 1MB
  int*   rowIdx   = (int*)((char*)d_ws + (size_t)BH * BUCKETS * BUCKETS * 4);
  float* rowScale = (float*)((char*)rowIdx + (size_t)BH * BUCKETS * 4);

  sortnet_kernel<<<dim3(BUCKETS, BH), 128, 0, stream>>>(q, k, W, ug, rbuf);

  const size_t sh_sink = (size_t)BUCKETS * 129 * sizeof(float);    // 66,048 B
  sinkhorn_kernel<<<BH, 128, sh_sink, stream>>>(rbuf, rowIdx, rowScale);

  const size_t sh_attn = (size_t)(64 * LDK + 128 * LDK + 128 * LDK) * sizeof(__bf16)
                       + (size_t)(64 * LDA) * sizeof(float);       // 120,832 B (< 320KB WGP LDS)
  attn_kernel<<<dim3(BUCKETS, BH), 256, sh_attn, stream>>>(q, k, v, rowIdx, rowScale, out);
}
